// MLA_5523327943226
// MI455X (gfx1250) — matrix operand order repaired and checked by reading, not compiled, not run on hardware
//
#include <hip/hip_runtime.h>
#include <math.h>

// ---------------------------------------------------------------------------
// MLA forward for MI455X (gfx1250).  bf16 WMMA (v_wmma_f32_16x16x32_bf16)
// with f32 accumulation; f32 softmax / RMSNorm / RoPE.
// Shapes: B=1, T=2048, DIM=2048, H=16, NOPE=128, ROPE=64, QK=192,
//         KV_RANK=512, VDIM=128.
// ---------------------------------------------------------------------------

typedef __attribute__((ext_vector_type(16))) __bf16 v16bf;
typedef __attribute__((ext_vector_type(2)))  __bf16 v2bf;
typedef __attribute__((ext_vector_type(8)))  float  v8f;

#define T_SEQ   2048
#define DIM     2048
#define NHEAD   16
#define NOPE    128
#define ROPE_D  64
#define QKD     192      // NOPE + ROPE
#define KV_RANK 512
#define VDIM    128
#define KVB_N   (NHEAD * (NOPE + VDIM))   // 4096
#define Q_N     (NHEAD * QKD)             // 3072
#define KVA_N   (KV_RANK + ROPE_D)        // 576
#define ATTN_SCALE 0.07216878364870322f   // 192^-0.5

// CDNA5 WMMA 16-bit A-operand layout: pair p (p=0..7) of v16bf holds
// K = (p<4 ? 2p : 16+2(p-4)) + half*8  (half = lane>=16), row M = lane%16.
__device__ __forceinline__ int kpairA(int p, int half) {
  return (p < 4 ? 2 * p : 16 + 2 * (p - 4)) + half * 8;
}
// B-operand: pair p holds K = (p<4 ? 2p : 16+2(p-4)) + half*8, column N = lane%16.
__device__ __forceinline__ int kpairB(int p, int half) {
  return (p < 4 ? 2 * p : 16 + 2 * (p - 4)) + half * 8;
}

// ---------------------------------------------------------------------------
// Generic GEMM:  C[M,N] = A[M,K] * B[N,K]^T   (fp32 in/out, bf16 WMMA math)
// Block: 128 threads (4 waves), tile 128(M) x 64(N), K step 32.
// Wave w owns rows [w*32, w*32+32) x all 64 cols: 2x4 WMMA tiles -> 8 WMMA
// per K-chunk from 6 operand builds (each build = 2x ds_load_b128).
// ---------------------------------------------------------------------------
#define GLD 40   // LDS row stride in bf16 (80B -> conflict-free 16-lane gather)

__global__ __launch_bounds__(128) void gemm_nt_kernel(
    const float* __restrict__ A, const float* __restrict__ B,
    float* __restrict__ C, int M, int N, int K)
{
  __shared__ __bf16 as[128 * GLD];   // 10240 B
  __shared__ __bf16 bs[64 * GLD];    //  5120 B

  const int tid  = threadIdx.x;
  const int lane = tid & 31;
  const int wid  = tid >> 5;
  const int half = (lane >> 4) & 1;
  const int lc   = lane & 15;
  const int m0   = blockIdx.y * 128;
  const int n0   = blockIdx.x * 64;
  const int wm   = wid * 32;          // wave row offset in tile

  v8f acc[2][4];
#pragma unroll
  for (int i = 0; i < 2; ++i)
#pragma unroll
    for (int j = 0; j < 4; ++j)
#pragma unroll
      for (int r = 0; r < 8; ++r) acc[i][j][r] = 0.0f;

  for (int k0 = 0; k0 < K; k0 += 32) {
    // Stage A (128x32) and B (64x32) fp32 tiles into LDS as bf16.
#pragma unroll
    for (int i = 0; i < 8; ++i) {
      int e  = tid + i * 128;          // 0..1023 float4 slots
      int r  = e >> 3;
      int c4 = (e & 7) << 2;
      float4 fa = *(const float4*)(A + (size_t)(m0 + r) * K + k0 + c4);
      __bf16* da = &as[r * GLD + c4];
      da[0] = (__bf16)fa.x; da[1] = (__bf16)fa.y;
      da[2] = (__bf16)fa.z; da[3] = (__bf16)fa.w;
    }
#pragma unroll
    for (int i = 0; i < 4; ++i) {
      int e  = tid + i * 128;          // 0..511 float4 slots
      int r  = e >> 3;
      int c4 = (e & 7) << 2;
      float4 fb = *(const float4*)(B + (size_t)(n0 + r) * K + k0 + c4);
      __bf16* db = &bs[r * GLD + c4];
      db[0] = (__bf16)fb.x; db[1] = (__bf16)fb.y;
      db[2] = (__bf16)fb.z; db[3] = (__bf16)fb.w;
    }
    __syncthreads();

    v16bf av[2], bv[4];
#pragma unroll
    for (int s = 0; s < 2; ++s) {
      int row = wm + s * 16 + lc;
#pragma unroll
      for (int p = 0; p < 8; ++p) {
        v2bf pr = *(const v2bf*)&as[row * GLD + kpairA(p, half)];
        av[s][2 * p] = pr[0]; av[s][2 * p + 1] = pr[1];
      }
    }
#pragma unroll
    for (int j = 0; j < 4; ++j) {
      int col = j * 16 + lc;
#pragma unroll
      for (int p = 0; p < 8; ++p) {
        v2bf pr = *(const v2bf*)&bs[col * GLD + kpairB(p, half)];
        bv[j][2 * p] = pr[0]; bv[j][2 * p + 1] = pr[1];
      }
    }
#pragma unroll
    for (int i = 0; i < 2; ++i)
#pragma unroll
      for (int j = 0; j < 4; ++j)
        acc[i][j] = __builtin_amdgcn_wmma_f32_16x16x32_bf16(
            false, av[i], false, bv[j], (short)0, acc[i][j], false, false);
    __syncthreads();
  }

  // C layout: reg r -> row (r + 8*half), col lane%16.
#pragma unroll
  for (int i = 0; i < 2; ++i)
#pragma unroll
    for (int j = 0; j < 4; ++j)
#pragma unroll
      for (int r = 0; r < 8; ++r) {
        int row = m0 + wm + i * 16 + r + half * 8;
        int col = n0 + j * 16 + lc;
        C[(size_t)row * N + col] = acc[i][j][r];
      }
}

// ---------------------------------------------------------------------------
// RoPE: q_pe (T,H,64 tail of 192) and k_pe (kv[:,512:576]) in place.
// One block per t, 544 threads = 512 (q pairs) + 32 (k_pe pairs).
// ---------------------------------------------------------------------------
__global__ __launch_bounds__(544) void rope_kernel(
    float* __restrict__ q, float* __restrict__ kv,
    const float* __restrict__ cosb, const float* __restrict__ sinb)
{
  const int t = blockIdx.x;
  const int j = threadIdx.x;
  float* p;
  int i;
  if (j < NHEAD * 32) {
    int hh = j >> 5; i = j & 31;
    p = q + (size_t)t * Q_N + hh * QKD + NOPE + 2 * i;
  } else {
    i = j - NHEAD * 32;
    p = kv + (size_t)t * KVA_N + KV_RANK + 2 * i;
  }
  float c = cosb[t * 32 + i], s = sinb[t * 32 + i];
  float xe = p[0], xo = p[1];
  p[0] = xe * c - xo * s;
  p[1] = xe * s + xo * c;
}

// ---------------------------------------------------------------------------
// RMSNorm of kv latent (512 of 576) with weight. One block per row.
// ---------------------------------------------------------------------------
__global__ __launch_bounds__(256) void rms_kernel(
    const float* __restrict__ kv, const float* __restrict__ w,
    float* __restrict__ outn)
{
  __shared__ float red[8];
  const int t = blockIdx.x;
  const float* row = kv + (size_t)t * KVA_N;
  float ss = 0.0f;
  for (int i = threadIdx.x; i < KV_RANK; i += 256) {
    float v = row[i]; ss += v * v;
  }
#pragma unroll
  for (int off = 16; off > 0; off >>= 1) ss += __shfl_xor(ss, off);
  if ((threadIdx.x & 31) == 0) red[threadIdx.x >> 5] = ss;
  __syncthreads();
  float tot = 0.0f;
#pragma unroll
  for (int i = 0; i < 8; ++i) tot += red[i];
  float scale = rsqrtf(tot * (1.0f / (float)KV_RANK) + 1e-6f);
  for (int i = threadIdx.x; i < KV_RANK; i += 256)
    outn[(size_t)t * KV_RANK + i] = row[i] * scale * w[i];
}

// ---------------------------------------------------------------------------
// Flash attention, one head per blockIdx.y, 64 query rows per block
// (4 waves x 16 rows), key blocks of 64, causal, online softmax.
//   q   : (T, H*192)  post-RoPE
//   kvb : (T, H*256)  [k_nope | v] per head
//   kvr : (T, 576)    rope'd k_pe in cols 512..575
//   out : (T, H*128)
// ---------------------------------------------------------------------------
#define KS_LD 200   // bf16 stride for K tile rows (key-major, 192 feats)
#define VT_LD 72    // bf16 stride for V^T rows (dim-major, 64 keys)
#define PS_LD 72    // bf16 stride for per-wave P tile rows

__global__ __launch_bounds__(128) void mla_attn_kernel(
    const float* __restrict__ q, const float* __restrict__ kvb,
    const float* __restrict__ kvr, float* __restrict__ out)
{
  __shared__ __bf16 ks[64 * KS_LD];          // 25600 B
  __shared__ __bf16 vts[VDIM * VT_LD];       // 18432 B
  __shared__ __bf16 ps[4 * 16 * PS_LD];      //  9216 B

  const int tid   = threadIdx.x;
  const int lane  = tid & 31;
  const int wid   = tid >> 5;
  const int half  = (lane >> 4) & 1;
  const int lc    = lane & 15;
  const int h     = blockIdx.y;
  const int qb    = blockIdx.x;            // query block of 64 rows
  const int qrow0 = qb * 64 + wid * 16;    // this wave's first query row
  const int myrow = qrow0 + lc;            // A-operand row for this lane

  // Hoist Q operands (6 K-chunks of 32) straight from global into registers.
  v16bf qa[6];
  const float* qp = q + (size_t)myrow * Q_N + h * QKD;
#pragma unroll
  for (int kc = 0; kc < 6; ++kc)
#pragma unroll
    for (int p = 0; p < 8; ++p) {
      int kk = kc * 32 + kpairA(p, half);
      float2 f = *(const float2*)(qp + kk);
      qa[kc][2 * p] = (__bf16)f.x; qa[kc][2 * p + 1] = (__bf16)f.y;
    }

  v8f o[8];
  float m[8], l[8];
#pragma unroll
  for (int vt = 0; vt < 8; ++vt)
#pragma unroll
    for (int r = 0; r < 8; ++r) o[vt][r] = 0.0f;
#pragma unroll
  for (int r = 0; r < 8; ++r) { m[r] = -3.0e38f; l[r] = 0.0f; }

  for (int kb = 0; kb <= qb; ++kb) {
    const int k0 = kb * 64;
    __syncthreads();   // previous iteration's consumers are done

    // Stage K tile: 64 keys x 192 feats (nope from kvb, rope from kvr).
    for (int i = 0; i < 96; ++i) {
      int e = tid + i * 128;
      int key = e / QKD, f = e - key * QKD;
      float val = (f < NOPE)
          ? kvb[(size_t)(k0 + key) * KVB_N + h * 256 + f]
          : kvr[(size_t)(k0 + key) * KVA_N + KV_RANK + (f - NOPE)];
      ks[key * KS_LD + f] = (__bf16)val;
    }
    // Stage V^T tile: 128 dims x 64 keys.
    for (int i = 0; i < 64; ++i) {
      int e = tid + i * 128;
      int n = e & 127, kk = e >> 7;
      vts[n * VT_LD + kk] =
          (__bf16)kvb[(size_t)(k0 + kk) * KVB_N + h * 256 + NOPE + n];
    }
    __syncthreads();

    // S = Q K^T  (4 N-tiles x 6 K-chunks of WMMA)
    v8f s[4];
#pragma unroll
    for (int nt = 0; nt < 4; ++nt) {
#pragma unroll
      for (int r = 0; r < 8; ++r) s[nt][r] = 0.0f;
      int col = nt * 16 + lc;
#pragma unroll
      for (int kc = 0; kc < 6; ++kc) {
        v16bf kv_op;
#pragma unroll
        for (int p = 0; p < 8; ++p) {
          v2bf pr = *(const v2bf*)&ks[col * KS_LD + kc * 32 + kpairB(p, half)];
          kv_op[2 * p] = pr[0]; kv_op[2 * p + 1] = pr[1];
        }
        s[nt] = __builtin_amdgcn_wmma_f32_16x16x32_bf16(
            false, qa[kc], false, kv_op, (short)0, s[nt], false, false);
      }
    }

    // Scale + causal mask (diagonal block only).
#pragma unroll
    for (int nt = 0; nt < 4; ++nt)
#pragma unroll
      for (int r = 0; r < 8; ++r) {
        float v = s[nt][r] * ATTN_SCALE;
        if (kb == qb) {
          int key = k0 + nt * 16 + lc;
          int qr  = qrow0 + r + half * 8;
          if (key > qr) v = -3.0e38f;
        }
        s[nt][r] = v;
      }

    // Online softmax: 16-lane row reductions (rows live on lane%16 groups).
    float alpha[8];
#pragma unroll
    for (int r = 0; r < 8; ++r) {
      float mx = fmaxf(fmaxf(s[0][r], s[1][r]), fmaxf(s[2][r], s[3][r]));
#pragma unroll
      for (int off = 1; off <= 8; off <<= 1) mx = fmaxf(mx, __shfl_xor(mx, off));
      float mn = fmaxf(m[r], mx);
      alpha[r] = __expf(m[r] - mn);
      m[r] = mn;
    }
#pragma unroll
    for (int nt = 0; nt < 4; ++nt)
#pragma unroll
      for (int r = 0; r < 8; ++r) s[nt][r] = __expf(s[nt][r] - m[r]);
#pragma unroll
    for (int r = 0; r < 8; ++r) {
      float rs = s[0][r] + s[1][r] + s[2][r] + s[3][r];
#pragma unroll
      for (int off = 1; off <= 8; off <<= 1) rs += __shfl_xor(rs, off);
      l[r] = l[r] * alpha[r] + rs;
    }
#pragma unroll
    for (int vt = 0; vt < 8; ++vt)
#pragma unroll
      for (int r = 0; r < 8; ++r) o[vt][r] *= alpha[r];

    // Re-layout P via per-wave LDS scratch into A-operand format.
    const int psb = wid * 16 * PS_LD;
#pragma unroll
    for (int nt = 0; nt < 4; ++nt)
#pragma unroll
      for (int r = 0; r < 8; ++r)
        ps[psb + (r + half * 8) * PS_LD + nt * 16 + lc] = (__bf16)s[nt][r];
    __syncthreads();   // also guarantees wave-visible P before gather

    v16bf pa[2];
#pragma unroll
    for (int ch = 0; ch < 2; ++ch)
#pragma unroll
      for (int p = 0; p < 8; ++p) {
        v2bf pr = *(const v2bf*)&ps[psb + lc * PS_LD + ch * 32 + kpairA(p, half)];
        pa[ch][2 * p] = pr[0]; pa[ch][2 * p + 1] = pr[1];
      }

    // O += P V  (8 dim-tiles x 2 K-chunks of 32 keys)
#pragma unroll
    for (int vt = 0; vt < 8; ++vt) {
      int col = vt * 16 + lc;
#pragma unroll
      for (int ch = 0; ch < 2; ++ch) {
        v16bf vb;
#pragma unroll
        for (int p = 0; p < 8; ++p) {
          v2bf pr = *(const v2bf*)&vts[col * VT_LD + ch * 32 + kpairB(p, half)];
          vb[2 * p] = pr[0]; vb[2 * p + 1] = pr[1];
        }
        o[vt] = __builtin_amdgcn_wmma_f32_16x16x32_bf16(
            false, pa[ch], false, vb, (short)0, o[vt], false, false);
      }
    }
  }

  // Normalize and write out (T, H*128).
  float linv[8];
#pragma unroll
  for (int r = 0; r < 8; ++r) linv[r] = 1.0f / l[r];
#pragma unroll
  for (int vt = 0; vt < 8; ++vt)
#pragma unroll
    for (int r = 0; r < 8; ++r) {
      int row = qrow0 + r + half * 8;
      out[(size_t)row * (NHEAD * VDIM) + h * VDIM + vt * 16 + lc] =
          o[vt][r] * linv[r];
    }
}

// ---------------------------------------------------------------------------
// Host launcher
// ---------------------------------------------------------------------------
extern "C" void kernel_launch(void* const* d_in, const int* in_sizes, int n_in,
                              void* d_out, int out_size, void* d_ws, size_t ws_size,
                              hipStream_t stream)
{
  const float* x         = (const float*)d_in[0];
  const float* wq        = (const float*)d_in[1];
  const float* wkv_a     = (const float*)d_in[2];
  const float* kv_norm_w = (const float*)d_in[3];
  const float* wkv_b     = (const float*)d_in[4];
  const float* wo        = (const float*)d_in[5];
  const float* cosb      = (const float*)d_in[6];
  const float* sinb      = (const float*)d_in[7];

  float* ws   = (float*)d_ws;                 // ~84.4 MB of f32 scratch
  float* qbuf = ws;                           // 2048*3072
  float* kv   = qbuf + (size_t)T_SEQ * Q_N;   // 2048*576
  float* kvn  = kv   + (size_t)T_SEQ * KVA_N; // 2048*512
  float* kvbb = kvn  + (size_t)T_SEQ * KV_RANK; // 2048*4096
  float* attn = kvbb + (size_t)T_SEQ * KVB_N; // 2048*2048
  float* outp = (float*)d_out;

  // 1) q = x @ wq^T            (2048 x 3072, K=2048)
  gemm_nt_kernel<<<dim3(Q_N / 64, T_SEQ / 128), 128, 0, stream>>>(
      x, wq, qbuf, T_SEQ, Q_N, DIM);
  // 2) kv = x @ wkv_a^T        (2048 x 576, K=2048)
  gemm_nt_kernel<<<dim3(KVA_N / 64, T_SEQ / 128), 128, 0, stream>>>(
      x, wkv_a, kv, T_SEQ, KVA_N, DIM);
  // 3) RoPE on q_pe and k_pe (in place)
  rope_kernel<<<T_SEQ, 544, 0, stream>>>(qbuf, kv, cosb, sinb);
  // 4) RMSNorm of latent
  rms_kernel<<<T_SEQ, 256, 0, stream>>>(kv, kv_norm_w, kvn);
  // 5) kvb = norm(kv_lat) @ wkv_b^T   (2048 x 4096, K=512)
  gemm_nt_kernel<<<dim3(KVB_N / 64, T_SEQ / 128), 128, 0, stream>>>(
      kvn, wkv_b, kvbb, T_SEQ, KVB_N, KV_RANK);
  // 6) causal flash attention per head
  mla_attn_kernel<<<dim3(T_SEQ / 64, NHEAD), 128, 0, stream>>>(
      qbuf, kvbb, kv, attn);
  // 7) out = attn @ wo^T       (2048 x 2048, K=2048)
  gemm_nt_kernel<<<dim3(DIM / 64, T_SEQ / 128), 128, 0, stream>>>(
      attn, wo, outp, T_SEQ, DIM, NHEAD * VDIM);
}
